// FlowHmmOptim_28991029248642
// MI455X (gfx1250) — matrix operand order repaired and checked by reading, not compiled, not run on hardware
//
#include <hip/hip_runtime.h>

// ---------------------------------------------------------------------------
// FlowHmm on MI455X (gfx1250): WMMA f16 16x16x32 for the per-layer MLPs,
// async global->LDS DMA (ASYNCcnt) for double-buffered weight staging.
// ---------------------------------------------------------------------------

typedef _Float16 v16h __attribute__((ext_vector_type(16)));
typedef _Float16 v8h  __attribute__((ext_vector_type(8)));
typedef _Float16 v4h  __attribute__((ext_vector_type(4)));
typedef float    v8f  __attribute__((ext_vector_type(8)));

#define WAVE_DS_SYNC() asm volatile("s_wait_dscnt 0x0" ::: "memory")
#define WAIT_ASYNC()   asm volatile("s_wait_asynccnt 0x0" ::: "memory")

constexpr int FC = 8;      // components
constexpr int FK = 64;     // flow layers
constexpr int FD = 32;     // obs dim
constexpr int FN = 2048;   // grid nodes
constexpr float NEG_HALF_D_LOG_2PI = -29.4060330624f;  // -0.5*32*ln(2*pi)

// LDS byte offset of a __shared__ object (addrspace(3) offset).
static __device__ __forceinline__ unsigned lds_off(const void* p) {
  return (unsigned)(unsigned long long)(
      const __attribute__((address_space(3))) char*)p;
}

// Async DMA: 16 bytes / lane, memory -> LDS, tracked with ASYNCcnt.
static __device__ __forceinline__ void async_b128(unsigned ldsByteAddr,
                                                  const void* gaddr) {
  asm volatile("global_load_async_to_lds_b128 %0, %1, off"
               :
               : "v"(ldsByteAddr), "v"((unsigned long long)gaddr)
               : "memory");
}

static __device__ __forceinline__ v8f wmma16(v16h a, v16h b, v8f c) {
  // D = A(16x32,f16) * B(32x16,f16) + C(16x16,f32)
  return __builtin_amdgcn_wmma_f32_16x16x32_f16(false, a, false, b, (short)0, c,
                                                false, false);
}

// Load a 16x32 f16 A-fragment from LDS stored row-major [M][k].
// Lane L holds row M=L%16; halves 0..15 -> k = kLow..kLow+15,
// with kLow = (L>=16 ? 16 : 0).
static __device__ __forceinline__ v16h loadAfrag(const _Float16* p, int base) {
  v8h lo = *reinterpret_cast<const v8h*>(p + base);
  v8h hh = *reinterpret_cast<const v8h*>(p + base + 8);
  v16h a;
#pragma unroll
  for (int i = 0; i < 8; ++i) { a[i] = lo[i]; a[i + 8] = hh[i]; }
  return a;
}

// One MLP: out[2] (f32 C-tiles for output features 0-15 / 16-31)
// W1 in LDS: [64 out][32 in] row-major; W2 in LDS: [32 out][64 in] row-major
static __device__ __forceinline__ void run_mlp(v16h a,
                                               const _Float16* W1,
                                               const _Float16* W2,
                                               const float* __restrict__ gb1,
                                               const float* __restrict__ gb2,
                                               _Float16* hb,
                                               int m, int hi, v8f out[2]) {
  // ---- layer 1: H = relu(A * W1^T + b1), H is 16x64 -----------------------
#pragma unroll
  for (int ht = 0; ht < 4; ++ht) {
    const int n = ht * 16 + m;
    v16h b = *reinterpret_cast<const v16h*>(W1 + n * 32 + hi * 16);
    v8f c = {};
    c = wmma16(a, b, c);
    const float bias = gb1[n];
#pragma unroll
    for (int r = 0; r < 8; ++r) {
      float v = c[r] + bias;
      v = v > 0.f ? v : 0.f;                         // ReLU
      hb[(r + hi * 8) * 64 + n] = (_Float16)v;       // stage H for transpose
    }
  }
  WAVE_DS_SYNC();
  // ---- A-fragments of H (K=64 -> two chained 16x16x32 WMMAs) --------------
  const int hbase = m * 64 + hi * 16;
  v16h a0 = loadAfrag(hb, hbase);
  v16h a1 = loadAfrag(hb, hbase + 32);
  // ---- layer 2: OUT = H * W2^T + b2, OUT is 16x32 -------------------------
#pragma unroll
  for (int ot = 0; ot < 2; ++ot) {
    const int n = ot * 16 + m;
    v16h b0 = *reinterpret_cast<const v16h*>(W2 + n * 64 + hi * 16);
    v16h b1 = *reinterpret_cast<const v16h*>(W2 + n * 64 + 32 + hi * 16);
    v8f c = {};
    c = wmma16(a0, b0, c);
    c = wmma16(a1, b1, c);
    const float bias = gb2[n];
#pragma unroll
    for (int r = 0; r < 8; ++r) c[r] += bias;
    out[ot] = c;
  }
}

// ---------------------------------------------------------------------------
// One-time weight conversion: f32 [4 arrays][C*K][2048] -> f16, layer-major
// [(c*K+k)][x][2048] so each layer's 4 matrices are 16KB contiguous.
// ---------------------------------------------------------------------------
__global__ __launch_bounds__(256) void k_wcvt(const float* __restrict__ s1,
                                              const float* __restrict__ s2,
                                              const float* __restrict__ t1,
                                              const float* __restrict__ t2,
                                              _Float16* __restrict__ dst) {
  const float* src[4] = {s1, s2, t1, t2};
  const int e4 = (blockIdx.x * 256 + threadIdx.x) * 4;  // 4M elements total
  const int i  = e4 & 2047;
  const int x  = (e4 >> 11) & 3;
  const int ck = e4 >> 13;
  float4 v = *reinterpret_cast<const float4*>(src[x] + (size_t)ck * 2048 + i);
  v4h h;
  h[0] = (_Float16)v.x; h[1] = (_Float16)v.y;
  h[2] = (_Float16)v.z; h[3] = (_Float16)v.w;
  *reinterpret_cast<v4h*>(dst + (((size_t)ck * 4 + x) << 11) + i) = h;
}

// ---------------------------------------------------------------------------
// Flow kernel: 8 waves per block, each wave owns a 16-node tile; iterates the
// K=64 layers in reverse. Weights staged in LDS per layer:
//  - preconv path: async DMA (global_load_async_to_lds_b128), double-buffered
//  - fallback path: f32 loads + cvt + ds_store (single buffer)
// ---------------------------------------------------------------------------
__global__ __launch_bounds__(256) void k_flow(
    const float* __restrict__ nodes,
    const float* __restrict__ sW1, const float* __restrict__ sb1,
    const float* __restrict__ sW2, const float* __restrict__ sb2,
    const float* __restrict__ tW1, const float* __restrict__ tb1,
    const float* __restrict__ tW2, const float* __restrict__ tb2,
    const float* __restrict__ an_s, const float* __restrict__ an_t,
    const float* __restrict__ loc, const float* __restrict__ lsc,
    const _Float16* __restrict__ wsW,   // preconverted f16 weights (or null)
    int preconv,
    float* __restrict__ logp) {
  __shared__ _Float16 Wlds[2][4][2048];  // double-buffered weights     32KB
  __shared__ _Float16 Zm[8][16 * 32];    // per-wave masked-z staging    8KB
  __shared__ _Float16 Hb[8][16 * 64];    // per-wave hidden staging     16KB

  const int tid  = threadIdx.x;
  const int lane = tid & 31;
  const int wave = tid >> 5;
  const int m    = lane & 15;
  const int hi   = (lane >= 16) ? 1 : 0;

  const int c        = blockIdx.x >> 4;
  const int nodeBase = (blockIdx.x & 15) * 128 + wave * 16;

  // z in WMMA C-layout: z[t][r] = feature (t*16 + m) of node (nodeBase+r+hi*8)
  float z[2][8], ldp[2][8];
#pragma unroll
  for (int t = 0; t < 2; ++t) {
    const int f = t * 16 + m;
#pragma unroll
    for (int r = 0; r < 8; ++r) {
      z[t][r] = nodes[(nodeBase + r + hi * 8) * FD + f];
      ldp[t][r] = 0.f;
    }
  }

  const float* gW[4] = {sW1, sW2, tW1, tW2};
  _Float16* zwave = &Zm[wave][0];
  _Float16* hwave = &Hb[wave][0];

  // Prologue: kick off async DMA for the first (reversed) layer.
  if (preconv) {
    const _Float16* gbase = wsW + ((size_t)c * FK + (FK - 1)) * 8192;
    const unsigned lbase = lds_off(&Wlds[(FK - 1) & 1][0][0]);
#pragma unroll
    for (int x = 0; x < 4; ++x)
      async_b128(lbase + x * 4096 + tid * 16, gbase + x * 2048 + tid * 8);
  }

  for (int kk = FK - 1; kk >= 0; --kk) {
    const size_t off = ((size_t)c * FK + kk) * 2048;
    const _Float16* Wcur;
    if (preconv) {
      WAIT_ASYNC();        // this wave's DMA for layer kk has landed
      __syncthreads();     // ...and every other wave's too; prev reads done
      const int buf = kk & 1;
      if (kk > 0) {        // overlap: DMA next layer into the other buffer
        const _Float16* gbase = wsW + ((size_t)c * FK + (kk - 1)) * 8192;
        const unsigned lbase = lds_off(&Wlds[buf ^ 1][0][0]);
#pragma unroll
        for (int x = 0; x < 4; ++x)
          async_b128(lbase + x * 4096 + tid * 16, gbase + x * 2048 + tid * 8);
      }
      Wcur = &Wlds[buf][0][0];
    } else {
      __syncthreads();     // previous layer's weights fully consumed
#pragma unroll
      for (int x = 0; x < 4; ++x) {
        const float* g = gW[x] + off + tid * 8;
        float4 p0 = *reinterpret_cast<const float4*>(g);
        float4 p1 = *reinterpret_cast<const float4*>(g + 4);
        v8h hv;
        hv[0] = (_Float16)p0.x; hv[1] = (_Float16)p0.y;
        hv[2] = (_Float16)p0.z; hv[3] = (_Float16)p0.w;
        hv[4] = (_Float16)p1.x; hv[5] = (_Float16)p1.y;
        hv[6] = (_Float16)p1.z; hv[7] = (_Float16)p1.w;
        *reinterpret_cast<v8h*>(&Wlds[0][x][tid * 8]) = hv;
        if (kk > 0) __builtin_prefetch(g - 2048, 0, 0);  // global_prefetch_b8
      }
      __syncthreads();
      Wcur = &Wlds[0][0][0];
    }

    // ---- ActNorm inverse + checkerboard mask ------------------------------
    float asv[2], atv[2];
    int mk[2];
#pragma unroll
    for (int t = 0; t < 2; ++t) {
      const int f = t * 16 + m;
      asv[t] = an_s[((size_t)c * FK + kk) * FD + f];
      atv[t] = an_t[((size_t)c * FK + kk) * FD + f];
      mk[t] = (((f + kk) & 1) == 0) ? 1 : 0;  // masks[kk][f]
      const float eas = __expf(-asv[t]);
#pragma unroll
      for (int r = 0; r < 8; ++r) {
        z[t][r] = (z[t][r] - atv[t]) * eas;
        ldp[t][r] -= asv[t];                  // ld -= sum(a_s) (per-feature)
        zwave[(r + hi * 8) * 32 + f] = (_Float16)(mk[t] ? z[t][r] : 0.f);
      }
    }
    WAVE_DS_SYNC();

    // ---- shared A-fragment of zm, then s- and t-MLPs ----------------------
    v16h a = loadAfrag(zwave, m * 32 + hi * 16);
    const float* gb1s = sb1 + ((size_t)c * FK + kk) * 64;
    const float* gb2s = sb2 + ((size_t)c * FK + kk) * 32;
    const float* gb1t = tb1 + ((size_t)c * FK + kk) * 64;
    const float* gb2t = tb2 + ((size_t)c * FK + kk) * 32;
    v8f sOut[2], tOut[2];
    run_mlp(a, Wcur + 0 * 2048, Wcur + 1 * 2048, gb1s, gb2s, hwave, m, hi,
            sOut);
    run_mlp(a, Wcur + 2 * 2048, Wcur + 3 * 2048, gb1t, gb2t, hwave, m, hi,
            tOut);

    // ---- coupling inverse: z = zm + (1-m)*(z - t)*exp(-s) -----------------
#pragma unroll
    for (int t = 0; t < 2; ++t) {
#pragma unroll
      for (int r = 0; r < 8; ++r) {
        if (!mk[t]) {
          const float sv = sOut[t][r];
          z[t][r] = (z[t][r] - tOut[t][r]) * __expf(-sv);
          ldp[t][r] -= sv;                    // ld -= sum((1-m)*s)
        }
      }
    }
  }

  // ---- DiagGaussian log_prob ----------------------------------------------
#pragma unroll
  for (int t = 0; t < 2; ++t) {
    const int f = t * 16 + m;
    const float lc = loc[c * FD + f];
    const float ls = lsc[c * FD + f];
    const float ei = __expf(-ls);
#pragma unroll
    for (int r = 0; r < 8; ++r) {
      const float u = (z[t][r] - lc) * ei;
      ldp[t][r] -= ls + 0.5f * u * u;
    }
  }
  // per-row sum over 32 features: butterfly within each 16-lane half
  float p[8];
#pragma unroll
  for (int r = 0; r < 8; ++r) {
    p[r] = ldp[0][r] + ldp[1][r];
#pragma unroll
    for (int sft = 8; sft >= 1; sft >>= 1) p[r] += __shfl_xor(p[r], sft, 16);
  }
  if (m == 0) {
#pragma unroll
    for (int r = 0; r < 8; ++r)
      logp[c * FN + nodeBase + hi * 8 + r] = p[r] + NEG_HALF_D_LOG_2PI;
  }
}

// ---------------------------------------------------------------------------
// rowsum[c] = max(sum_n exp(logp[c][n]), 1e-12)
// ---------------------------------------------------------------------------
__global__ __launch_bounds__(256) void k_rowsum(const float* __restrict__ logp,
                                                float* __restrict__ rs) {
  __shared__ float sb[256];
  const int c = blockIdx.x, tid = threadIdx.x;
  float acc = 0.f;
  for (int n = tid; n < FN; n += 256) acc += __expf(logp[c * FN + n]);
  sb[tid] = acc;
  __syncthreads();
  for (int s = 128; s > 0; s >>= 1) {
    if (tid < s) sb[tid] += sb[tid + s];
    __syncthreads();
  }
  if (tid == 0) rs[c] = fmaxf(sb[0], 1e-12f);
}

// ---------------------------------------------------------------------------
// Bm[c][n] = exp(logp)/rs[c];  A[c][n] = sum_c2 S[c][c2]*Bm[c2][n]
// ---------------------------------------------------------------------------
__global__ __launch_bounds__(256) void k_mix(const float* __restrict__ logp,
                                             const float* __restrict__ rs,
                                             const float* __restrict__ Sunc,
                                             float* __restrict__ Bm,
                                             float* __restrict__ A) {
  const int idx = blockIdx.x * 256 + threadIdx.x;
  const int c = idx >> 11, n = idx & (FN - 1);
  float ssum = 0.f;
#pragma unroll
  for (int i = 0; i < FC * FC; ++i) ssum += __expf(Sunc[i]);
  Bm[idx] = __expf(logp[idx]) / rs[c];
  float acc = 0.f;
#pragma unroll
  for (int c2 = 0; c2 < FC; ++c2)
    acc += (__expf(Sunc[c * FC + c2]) / ssum) * (__expf(logp[c2 * FN + n]) / rs[c2]);
  A[idx] = acc;
}

// ---------------------------------------------------------------------------
// out = Bm^T (N x 8) @ A (8 x N) -- bandwidth bound; 64x64 LDS tiles,
// 4x4 micro-tiles per thread, float4 stores.
// ---------------------------------------------------------------------------
__global__ __launch_bounds__(256) void k_outer(const float* __restrict__ Bm,
                                               const float* __restrict__ A,
                                               float* __restrict__ out) {
  __shared__ float Bs[FC][64], As[FC][64];
  const int tid = threadIdx.x;
  const int bi = blockIdx.y, bj = blockIdx.x;
#pragma unroll
  for (int l = 0; l < 2; ++l) {
    const int q = tid + l * 256;
    const int cc = q >> 6, e = q & 63;
    Bs[cc][e] = Bm[cc * FN + bi * 64 + e];
    As[cc][e] = A[cc * FN + bj * 64 + e];
  }
  __syncthreads();
  const int ti = tid >> 4, tj = tid & 15;
  float acc[4][4] = {};
#pragma unroll
  for (int cc = 0; cc < FC; ++cc) {
    float bv[4], av[4];
#pragma unroll
    for (int i = 0; i < 4; ++i) { bv[i] = Bs[cc][ti * 4 + i]; av[i] = As[cc][tj * 4 + i]; }
#pragma unroll
    for (int i = 0; i < 4; ++i)
#pragma unroll
      for (int j = 0; j < 4; ++j) acc[i][j] = fmaf(bv[i], av[j], acc[i][j]);
  }
#pragma unroll
  for (int i = 0; i < 4; ++i) {
    float4 vv = make_float4(acc[i][0], acc[i][1], acc[i][2], acc[i][3]);
    *reinterpret_cast<float4*>(
        &out[(size_t)(bi * 64 + ti * 4 + i) * FN + bj * 64 + tj * 4]) = vv;
  }
}

// ---------------------------------------------------------------------------
extern "C" void kernel_launch(void* const* d_in, const int* in_sizes, int n_in,
                              void* d_out, int out_size, void* d_ws,
                              size_t ws_size, hipStream_t stream) {
  (void)in_sizes; (void)n_in; (void)out_size;
  const float* nodes = (const float*)d_in[0];
  const float* sW1 = (const float*)d_in[1];
  const float* sb1 = (const float*)d_in[2];
  const float* sW2 = (const float*)d_in[3];
  const float* sb2 = (const float*)d_in[4];
  const float* tW1 = (const float*)d_in[5];
  const float* tb1 = (const float*)d_in[6];
  const float* tW2 = (const float*)d_in[7];
  const float* tb2 = (const float*)d_in[8];
  const float* an_s = (const float*)d_in[9];
  const float* an_t = (const float*)d_in[10];
  const float* loc = (const float*)d_in[11];
  const float* lsc = (const float*)d_in[12];
  const float* Sunc = (const float*)d_in[13];
  float* out = (float*)d_out;

  // f16 weight region: 4 matrices * C*K*2048 halves = 8 MiB, layer-major.
  const size_t WH = (size_t)4 * FC * FK * 2048;          // halves
  const size_t FLOATS = (size_t)FC * FN * 3 + 64;        // logp, Bm, A, rs
  const size_t NEED = WH * 2 + FLOATS * 4;
  const int preconv = (ws_size >= NEED) ? 1 : 0;

  _Float16* wsW = (_Float16*)d_ws;
  float* fscr = preconv ? (float*)((char*)d_ws + WH * 2) : (float*)d_ws;
  float* logp = fscr;                     // [C*N]
  float* rs   = fscr + FC * FN;           // [C]
  float* Bm   = fscr + FC * FN + 16;      // [C*N]
  float* A    = Bm + FC * FN;             // [C*N]

  if (preconv) {
    // one-time (per launch) f32 -> f16 weight conversion, layer-major
    k_wcvt<<<dim3((unsigned)(WH / 1024)), dim3(256), 0, stream>>>(
        sW1, sW2, tW1, tW2, wsW);
  }
  k_flow<<<dim3(FC * (FN / 128)), dim3(256), 0, stream>>>(
      nodes, sW1, sb1, sW2, sb2, tW1, tb1, tW2, tb2, an_s, an_t, loc, lsc,
      wsW, preconv, logp);
  k_rowsum<<<dim3(FC), dim3(256), 0, stream>>>(logp, rs);
  k_mix<<<dim3(FC * FN / 256), dim3(256), 0, stream>>>(logp, rs, Sunc, Bm, A);
  k_outer<<<dim3(FN / 64, FN / 64), dim3(256), 0, stream>>>(Bm, A, out);
}
